// ASTEmbeder_83717502534016
// MI455X (gfx1250) — matrix operand order repaired and checked by reading, not compiled, not run on hardware
//
#include <hip/hip_runtime.h>
#include <hip/hip_bf16.h>
#include <math.h>

#define HDIM 512
#define NB   256
#define SLEN 512
#define MARGIN_F 0.05f
#define EPS_F 1e-8f

typedef __bf16 bf16_t;
typedef __attribute__((ext_vector_type(16))) __bf16 v16bf;
typedef __attribute__((ext_vector_type(8)))  float  v8f;

union Frag {
    v16bf v;
    uint4 q[2];
};

// ---------- W convert + transpose: Wt[n*H + k] = bf16(W[k*H + n]) ----------
__global__ __launch_bounds__(256) void wconv_kernel(const float* __restrict__ W,
                                                    bf16_t* __restrict__ Wt) {
    int idx = blockIdx.x * 256 + threadIdx.x;          // idx = n*H + k
    int n = idx / HDIM;
    int k = idx % HDIM;
    Wt[idx] = (bf16_t)W[k * HDIM + n];
}

// ---------- serial prefix sum of segment counts (B=256, trivial) ----------
__global__ void offsets_kernel(const int* __restrict__ cnt, int* __restrict__ offs) {
    if (threadIdx.x == 0 && blockIdx.x == 0) {
        int a = 0;
        for (int b = 0; b < NB; ++b) { offs[b] = a; a += cnt[b]; }
        offs[NB] = a;
    }
}

// ---------- fused WMMA score kernel:  s = tanh( pre(x) @ W + b ) . v ----------
// pre(x) = tanh(x) for code mode, identity for desc mode.
// One block = 32 rows (two 16-row A tiles per wave; B frags reused x2).
__global__ __launch_bounds__(256) void score_kernel(
    const float* __restrict__ x, const bf16_t* __restrict__ Wt,
    const float* __restrict__ bias, const float* __restrict__ vvec,
    const float* __restrict__ vb, float* __restrict__ s_out, int applyTanh)
{
    __shared__ __align__(16) bf16_t ldsA[32][536];   // padded stride vs bank conflicts
    __shared__ float spart[8][32];                   // per-wave row partials
    const int tid = threadIdx.x;
    const long rowBase = (long)blockIdx.x * 32;

    // stage 32x512 fp32 tile -> bf16 LDS (tanh fused for code mode)
    {
        const int r  = tid >> 3;           // 0..31
        const int kb = (tid & 7) << 6;     // 0..448 step 64
        const float* src = x + (rowBase + r) * HDIM + kb;
        #pragma unroll
        for (int j = 0; j < 16; ++j) {
            float4 f = *reinterpret_cast<const float4*>(src + j * 4);
            if (applyTanh) {
                f.x = tanhf(f.x); f.y = tanhf(f.y);
                f.z = tanhf(f.z); f.w = tanhf(f.w);
            }
            ldsA[r][kb + j * 4 + 0] = (bf16_t)f.x;
            ldsA[r][kb + j * 4 + 1] = (bf16_t)f.y;
            ldsA[r][kb + j * 4 + 2] = (bf16_t)f.z;
            ldsA[r][kb + j * 4 + 3] = (bf16_t)f.w;
        }
    }
    __syncthreads();

    const int wave = tid >> 5;      // 0..7, each owns 64 output columns
    const int lane = tid & 31;
    const int lh   = lane & 15;
    const int hi   = lane >> 4;     // which K-half this lane carries
    const int n0w  = wave << 6;

    v8f acc[2][4] = {};
    for (int k0 = 0; k0 < HDIM; k0 += 32) {
        // A frags: lane M = lh (+16 for tile 1); lanes 0-15 carry K
        // [k0,k0+8)&[k0+16,k0+24), lanes 16-31 carry K [k0+8,k0+16)&[k0+24,k0+32)
        Frag a0, a1;
        {
            const bf16_t* ap0 = &ldsA[lh][k0 + hi * 8];
            a0.q[0] = *reinterpret_cast<const uint4*>(ap0);
            a0.q[1] = *reinterpret_cast<const uint4*>(ap0 + 16);
            const bf16_t* ap1 = &ldsA[lh + 16][k0 + hi * 8];
            a1.q[0] = *reinterpret_cast<const uint4*>(ap1);
            a1.q[1] = *reinterpret_cast<const uint4*>(ap1 + 16);
        }
        #pragma unroll
        for (int t = 0; t < 4; ++t) {
            // B frag: lane N = lh; same K split as A (lanes 16-31 hold K+8 / K+24)
            Frag bf;
            const bf16_t* bp = Wt + (size_t)(n0w + t * 16 + lh) * HDIM + k0 + hi * 8;
            bf.q[0] = *reinterpret_cast<const uint4*>(bp);
            bf.q[1] = *reinterpret_cast<const uint4*>(bp + 16);
            acc[0][t] = __builtin_amdgcn_wmma_f32_16x16x32_bf16(
                false, a0.v, false, bf.v, (short)0, acc[0][t], false, false);
            acc[1][t] = __builtin_amdgcn_wmma_f32_16x16x32_bf16(
                false, a1.v, false, bf.v, (short)0, acc[1][t], false, false);
        }
    }

    // epilogue: s_m = sum_n tanh(d[m][n] + b[n]) * v[n]
    // D layout: lane N = lh, VGPR r -> M = r + 8*hi (+16 for tile 1)
    float part[2][8];
    #pragma unroll
    for (int rt = 0; rt < 2; ++rt)
        #pragma unroll
        for (int r = 0; r < 8; ++r) part[rt][r] = 0.0f;
    #pragma unroll
    for (int t = 0; t < 4; ++t) {
        const int n  = n0w + t * 16 + lh;
        const float bn = bias[n];
        const float vn = vvec[n];
        #pragma unroll
        for (int rt = 0; rt < 2; ++rt)
            #pragma unroll
            for (int r = 0; r < 8; ++r)
                part[rt][r] += tanhf(acc[rt][t][r] + bn) * vn;
    }
    #pragma unroll
    for (int m = 1; m <= 8; m <<= 1) {
        #pragma unroll
        for (int rt = 0; rt < 2; ++rt)
            #pragma unroll
            for (int r = 0; r < 8; ++r)
                part[rt][r] += __shfl_xor(part[rt][r], m, 32);
    }
    if (lane == 0) {
        #pragma unroll
        for (int rt = 0; rt < 2; ++rt)
            #pragma unroll
            for (int r = 0; r < 8; ++r) spart[wave][rt * 16 + r] = part[rt][r];
    } else if (lane == 16) {
        #pragma unroll
        for (int rt = 0; rt < 2; ++rt)
            #pragma unroll
            for (int r = 0; r < 8; ++r) spart[wave][rt * 16 + 8 + r] = part[rt][r];
    }
    __syncthreads();
    if (tid < 32) {                       // fixed-order cross-wave sum: deterministic
        float s = vb[0];
        #pragma unroll
        for (int w = 0; w < 8; ++w) s += spart[w][tid];
        s_out[rowBase + tid] = s;
    }
}

// ---------- ragged segment softmax (code): overwrite s with weights ----------
__global__ __launch_bounds__(256) void seg_softmax_kernel(float* __restrict__ s,
                                                          const int* __restrict__ offs) {
    __shared__ float red[256];
    const int b = blockIdx.x, tid = threadIdx.x;
    const int base = offs[b];
    const int n = offs[b + 1] - base;
    float m = -3.0e38f;
    for (int i = tid; i < n; i += 256) m = fmaxf(m, s[base + i]);
    red[tid] = m; __syncthreads();
    for (int st = 128; st > 0; st >>= 1) {
        if (tid < st) red[tid] = fmaxf(red[tid], red[tid + st]);
        __syncthreads();
    }
    m = red[0]; __syncthreads();
    float sum = 0.0f;
    for (int i = tid; i < n; i += 256) {
        float e = expf(s[base + i] - m);
        s[base + i] = e;
        sum += e;
    }
    red[tid] = sum; __syncthreads();
    for (int st = 128; st > 0; st >>= 1) {
        if (tid < st) red[tid] += red[tid + st];
        __syncthreads();
    }
    const float inv = 1.0f / red[0];
    for (int i = tid; i < n; i += 256) s[base + i] *= inv;
}

// ---------- masked softmax over S (desc): weights, 0 beyond len ----------
__global__ __launch_bounds__(256) void desc_softmax_kernel(float* __restrict__ s,
                                                           const int* __restrict__ lens) {
    __shared__ float red[256];
    const int b = blockIdx.x, tid = threadIdx.x;
    const int base = b * SLEN;
    const int len = lens[b];
    float m = -3.0e38f;
    for (int i = tid; i < len; i += 256) m = fmaxf(m, s[base + i]);
    red[tid] = m; __syncthreads();
    for (int st = 128; st > 0; st >>= 1) {
        if (tid < st) red[tid] = fmaxf(red[tid], red[tid + st]);
        __syncthreads();
    }
    m = red[0]; __syncthreads();
    float sum = 0.0f;
    for (int i = tid; i < SLEN; i += 256) {
        float e = (i < len) ? expf(s[base + i] - m) : 0.0f;   // exp(-1e9-m) == 0 in fp32
        s[base + i] = e;
        sum += e;
    }
    red[tid] = sum; __syncthreads();
    for (int st = 128; st > 0; st >>= 1) {
        if (tid < st) red[tid] += red[tid + st];
        __syncthreads();
    }
    const float inv = 1.0f / red[0];
    for (int i = tid; i < SLEN; i += 256) s[base + i] *= inv;
}

// ---------- weighted pooling: out[b] = tanh( sum_i w_i * pre(x_i) ) ----------
__global__ __launch_bounds__(256) void pool_kernel(
    const float* __restrict__ x, const float* __restrict__ w,
    const int* __restrict__ offs, int fixedStride,
    float* __restrict__ out, int applyTanh)
{
    const int b = blockIdx.x;
    int base, n;
    if (offs) { base = offs[b]; n = offs[b + 1] - base; }
    else      { base = b * fixedStride; n = fixedStride; }
    const int c0 = threadIdx.x, c1 = threadIdx.x + 256;
    float a0 = 0.0f, a1 = 0.0f;
    for (int i = 0; i < n; ++i) {
        const float wi = w[base + i];
        if (wi == 0.0f) continue;                      // masked desc tokens
        const float* row = x + (size_t)(base + i) * HDIM;
        float u = row[c0], v = row[c1];
        if (applyTanh) { u = tanhf(u); v = tanhf(v); }
        a0 += wi * u; a1 += wi * v;
    }
    out[(size_t)b * HDIM + c0] = tanhf(a0);
    out[(size_t)b * HDIM + c1] = tanhf(a1);
}

// ---------- cosine sims + hinge loss mean ----------
__global__ __launch_bounds__(256) void loss_kernel(
    const float* __restrict__ cr, const float* __restrict__ ar,
    const float* __restrict__ nr, float* __restrict__ out)
{
    __shared__ float red[256];
    const int b = threadIdx.x;
    const float* c = cr + (size_t)b * HDIM;
    const float* a = ar + (size_t)b * HDIM;
    const float* g = nr + (size_t)b * HDIM;
    float dca = 0.f, dcn = 0.f, nc = 0.f, na = 0.f, nn = 0.f;
    for (int k = 0; k < HDIM; ++k) {
        const float cv = c[k], av = a[k], gv = g[k];
        dca += cv * av; dcn += cv * gv;
        nc += cv * cv; na += av * av; nn += gv * gv;
    }
    const float sa = dca / (sqrtf(nc) * sqrtf(na) + EPS_F);
    const float sn = dcn / (sqrtf(nc) * sqrtf(nn) + EPS_F);
    red[b] = fmaxf(MARGIN_F - sa + sn, 1e-6f);
    __syncthreads();
    for (int st = 128; st > 0; st >>= 1) {
        if (b < st) red[b] += red[b + st];
        __syncthreads();
    }
    if (b == 0) out[0] = red[0] / (float)NB;
}

extern "C" void kernel_launch(void* const* d_in, const int* in_sizes, int n_in,
                              void* d_out, int out_size, void* d_ws, size_t ws_size,
                              hipStream_t stream) {
    const float* all_node_h = (const float*)d_in[0];
    const int*   node_num   = (const int*)  d_in[1];
    const float* anc_feat   = (const float*)d_in[2];
    const int*   anc_len    = (const int*)  d_in[3];
    const float* neg_feat   = (const float*)d_in[4];
    const int*   neg_len    = (const int*)  d_in[5];
    const float* W1 = (const float*)d_in[6];
    const float* b1 = (const float*)d_in[7];
    const float* v1 = (const float*)d_in[8];
    const float* v1b = (const float*)d_in[9];
    const float* W2 = (const float*)d_in[10];
    const float* b2 = (const float*)d_in[11];
    const float* v2 = (const float*)d_in[12];
    const float* v2b = (const float*)d_in[13];

    const int rowsCode = in_sizes[0] / HDIM;   // 131072
    const int rowsDesc = in_sizes[2] / HDIM;   // 131072

    auto alignup = [](size_t x) { return (x + 255) & ~(size_t)255; };
    char* ws = (char*)d_ws;
    size_t off = 0;
    bf16_t* Wt1 = (bf16_t*)(ws + off); off = alignup(off + (size_t)HDIM * HDIM * sizeof(bf16_t));
    bf16_t* Wt2 = (bf16_t*)(ws + off); off = alignup(off + (size_t)HDIM * HDIM * sizeof(bf16_t));
    float* s_code = (float*)(ws + off); off = alignup(off + (size_t)rowsCode * sizeof(float));
    float* s_anc  = (float*)(ws + off); off = alignup(off + (size_t)rowsDesc * sizeof(float));
    float* s_neg  = (float*)(ws + off); off = alignup(off + (size_t)rowsDesc * sizeof(float));
    int*   offs   = (int*)  (ws + off); off = alignup(off + (size_t)(NB + 1) * sizeof(int));
    float* crep   = (float*)(ws + off); off = alignup(off + (size_t)NB * HDIM * sizeof(float));
    float* arep   = (float*)(ws + off); off = alignup(off + (size_t)NB * HDIM * sizeof(float));
    float* nrep   = (float*)(ws + off); off = alignup(off + (size_t)NB * HDIM * sizeof(float));

    // weight prep + segment offsets
    wconv_kernel<<<(HDIM * HDIM) / 256, 256, 0, stream>>>(W1, Wt1);
    wconv_kernel<<<(HDIM * HDIM) / 256, 256, 0, stream>>>(W2, Wt2);
    offsets_kernel<<<1, 32, 0, stream>>>(node_num, offs);

    // WMMA attention-logit GEMMs (bf16 in, f32 acc), fused tanh(.)·v epilogue
    score_kernel<<<rowsCode / 32, 256, 0, stream>>>(all_node_h, Wt1, b1, v1, v1b, s_code, 1);
    score_kernel<<<rowsDesc / 32, 256, 0, stream>>>(anc_feat,   Wt2, b2, v2, v2b, s_anc, 0);
    score_kernel<<<rowsDesc / 32, 256, 0, stream>>>(neg_feat,   Wt2, b2, v2, v2b, s_neg, 0);

    // softmaxes -> weights in place
    seg_softmax_kernel<<<NB, 256, 0, stream>>>(s_code, offs);
    desc_softmax_kernel<<<NB, 256, 0, stream>>>(s_anc, anc_len);
    desc_softmax_kernel<<<NB, 256, 0, stream>>>(s_neg, neg_len);

    // weighted pooling + outer tanh
    pool_kernel<<<NB, 256, 0, stream>>>(all_node_h, s_code, offs, 0,    crep, 1);
    pool_kernel<<<NB, 256, 0, stream>>>(anc_feat,   s_anc,  nullptr, SLEN, arep, 0);
    pool_kernel<<<NB, 256, 0, stream>>>(neg_feat,   s_neg,  nullptr, SLEN, nrep, 0);

    // cosine + hinge mean -> scalar
    loss_kernel<<<1, 256, 0, stream>>>(crep, arep, nrep, (float*)d_out);
}
